// bilstm_crf_44530220925204
// MI455X (gfx1250) — matrix operand order repaired and checked by reading, not compiled, not run on hardware
//
#include <hip/hip_runtime.h>
#include <hip/hip_bf16.h>

typedef __bf16 bf16;
typedef __attribute__((ext_vector_type(16))) bf16  v16bf;
typedef __attribute__((ext_vector_type(8)))  bf16  bf16x8;
typedef __attribute__((ext_vector_type(8)))  float v8f;

#define B_      64
#define S_      512
#define E_      50
#define H_      200
#define T_      17
#define KE_     64        // E padded to mult of 32
#define KH_     224       // H padded to mult of 32
#define K2H_    416       // 2H padded to mult of 32
#define NG_     1664      // per-(s,b) gate row stride (2*800 padded for tile overreach)
#define GROWS_  1600      // gate rows (both dirs)
#define WHROWS_ 832       // w_hh rows padded (tile-12 overreach up to row 807)
#define MROWS_  (B_*S_)   // 32768

// ---------------- fragment helpers (wave32 WMMA layouts) ----------------

__device__ __forceinline__ v16bf combine16(bf16x8 lo, bf16x8 hi) {
    union { bf16x8 p[2]; v16bf v; } u;
    u.p[0] = lo; u.p[1] = hi;
    return u.v;
}

// A-matrix 16x32 bf16 fragment. prow points at row start + k0 (row = tileM + (lane&15)).
// half = lane>>4: elements 0..7 = K[8*half .. +8), elements 8..15 = K[16+8*half .. +8)
__device__ __forceinline__ v16bf frag_a(const bf16* prow, int hf) {
    bf16x8 lo = *(const bf16x8*)(prow + 8 * hf);
    bf16x8 hi = *(const bf16x8*)(prow + 16 + 8 * hf);
    return combine16(lo, hi);
}

// B-matrix 32x16 bf16 fragment from W stored (N x K) row-major (i.e. B = W^T).
// prow points at W row (tileN + (lane&15)) + k0. Lane reads K[8*half .. +8) and K[16+8*half .. +8).
__device__ __forceinline__ v16bf frag_b(const bf16* prow, int hf) {
    const bf16* p = prow + 8 * hf;
    return combine16(*(const bf16x8*)p, *(const bf16x8*)(p + 16));
}

__device__ __forceinline__ v8f wmma_bf16(v16bf a, v16bf b, v8f c) {
    return __builtin_amdgcn_wmma_f32_16x16x32_bf16(false, a, false, b, (short)0, c, false, false);
}

__device__ __forceinline__ float sigmoidf_(float x) { return 1.0f / (1.0f + __expf(-x)); }

// ---------------- utility kernels ----------------

__global__ void zero_bf16_kernel(bf16* p, long n) {
    long t = (long)blockIdx.x * blockDim.x + threadIdx.x;
    if (t < n) p[t] = (bf16)0.0f;
}

__global__ void gather_emb_kernel(const int* __restrict__ x, const float* __restrict__ emb,
                                  bf16* __restrict__ dst) {
    long t = (long)blockIdx.x * blockDim.x + threadIdx.x;   // over MROWS_*KE_
    if (t >= (long)MROWS_ * KE_) return;
    int r = (int)(t >> 6), c = (int)(t & 63);
    int idx = x[r];
    float v = (c < E_ && idx != 0) ? emb[(size_t)idx * E_ + c] : 0.0f;   // PAD=0 row zeroed
    dst[t] = (bf16)v;
}

// copy (Rsrc x Csrc) fp32 -> (Rwrite x Cdst) bf16 at row offset, zero-padding
__global__ void pack_bf16_kernel(const float* __restrict__ src, bf16* __restrict__ dst,
                                 int Rsrc, int Csrc, int Rwrite, int Cdst, int rowOff) {
    long t = (long)blockIdx.x * blockDim.x + threadIdx.x;
    if (t >= (long)Rwrite * Cdst) return;
    int r = (int)(t / Cdst), c = (int)(t % Cdst);
    float v = (r < Rsrc && c < Csrc) ? src[(size_t)r * Csrc + c] : 0.0f;
    dst[(size_t)(rowOff + r) * Cdst + c] = (bf16)v;
}

__global__ void pack_bias_kernel(const float* __restrict__ bf_, const float* __restrict__ bb_,
                                 float* __restrict__ dst) {
    int t = blockIdx.x * blockDim.x + threadIdx.x;
    if (t >= NG_) return;
    float v = 0.0f;
    if (t < 800) v = bf_[t];
    else if (t < 1600) v = bb_[t - 800];
    dst[t] = v;
}

// ---------------- gate-precompute GEMM:  xg[s][b][:] = A(bf16) * W^T + bias ----------------
// A: (MROWS_ x K) bf16 row-major, row = b*S_+s.  W: (GROWS_ x K) bf16 row-major.
// Block = 8 waves; wave w owns one 16-row M tile x 64 N cols (4 accumulators).
__global__ __launch_bounds__(256)
void gates_gemm_kernel(const bf16* __restrict__ A, const bf16* __restrict__ W,
                       const float* __restrict__ bias, float* __restrict__ xg,
                       int K, int kTiles) {
    const int tid = threadIdx.x, w = tid >> 5, l = tid & 31;
    const int nc = l & 15, hf = l >> 4;
    const int mtile = blockIdx.x * 8 + w;
    const int nbase = blockIdx.y * 64;

    v8f acc[4] = {};
    const bf16* arow = A + (size_t)(mtile * 16 + nc) * K;
    for (int kt = 0; kt < kTiles; ++kt) {
        const int k0 = kt * 32;
        v16bf af = frag_a(arow + k0, hf);
#pragma unroll
        for (int j = 0; j < 4; ++j) {
            const bf16* brow = W + (size_t)(nbase + j * 16 + nc) * K + k0;
            acc[j] = wmma_bf16(af, frag_b(brow, hf), acc[j]);
        }
    }
    // C layout: lane holds col (l&15), rows 8*(l>>4)+v. Scatter (row=b*S+s) -> xg[s][b][col].
    const int row0 = mtile * 16 + 8 * hf;
    const int b = row0 >> 9, s0 = row0 & 511;
#pragma unroll
    for (int j = 0; j < 4; ++j) {
        const int col = nbase + j * 16 + nc;
        const float bia = bias[col];
        float* dst = xg + ((size_t)s0 * B_ + b) * NG_ + col;
#pragma unroll
        for (int v = 0; v < 8; ++v)
            dst[(size_t)v * B_ * NG_] = acc[j][v] + bia;
    }
}

// ---------------- persistent LSTM scan (one block per direction) ----------------
// 13 waves; wave w owns hidden tile w (units w*16..w*16+15). c stays in registers.
__global__ __launch_bounds__(416)
void lstm_scan_kernel(const float* __restrict__ xg, const bf16* __restrict__ whhF,
                      const bf16* __restrict__ whhB, bf16* __restrict__ out) {
    const int dir = blockIdx.x;
    const bf16* __restrict__ whh = dir ? whhB : whhF;
    const int tid = threadIdx.x, w = tid >> 5, l = tid & 31;
    const int nc = l & 15, hf = l >> 4;
    const int hunit = w * 16 + nc;

    __shared__ __align__(16) bf16 hbuf[2][64 * KH_];
    for (int i = tid; i < 2 * 64 * KH_; i += blockDim.x) (&hbuf[0][0])[i] = (bf16)0.0f;
    __syncthreads();

    float c_reg[4][8] = {};

    for (int t = 0; t < S_; ++t) {
        const int cur = t & 1, nxt = cur ^ 1;
        const int tt = dir ? (S_ - 1 - t) : t;
        const float* __restrict__ xgt = xg + (size_t)tt * B_ * NG_ + dir * 800;

        for (int mt = 0; mt < 4; ++mt) {
            v8f aI = {}, aF = {}, aG = {}, aO = {};
            const bf16* hrow = &hbuf[cur][(mt * 16 + nc) * KH_];
#pragma unroll
            for (int kt = 0; kt < 7; ++kt) {
                const int k0 = kt * 32;
                v16bf af = frag_a(hrow + k0, hf);
                aI = wmma_bf16(af, frag_b(whh + (size_t)(0 * 200 + hunit) * KH_ + k0, hf), aI);
                aF = wmma_bf16(af, frag_b(whh + (size_t)(1 * 200 + hunit) * KH_ + k0, hf), aF);
                aG = wmma_bf16(af, frag_b(whh + (size_t)(2 * 200 + hunit) * KH_ + k0, hf), aG);
                aO = wmma_bf16(af, frag_b(whh + (size_t)(3 * 200 + hunit) * KH_ + k0, hf), aO);
            }
            const int bbase = mt * 16 + 8 * hf;
#pragma unroll
            for (int v = 0; v < 8; ++v) {
                const int b = bbase + v;
                const float* xr = xgt + (size_t)b * NG_;
                float gi = aI[v] + xr[0 * 200 + hunit];
                float gf = aF[v] + xr[1 * 200 + hunit];
                float gg = aG[v] + xr[2 * 200 + hunit];
                float go = aO[v] + xr[3 * 200 + hunit];
                float c  = sigmoidf_(gf) * c_reg[mt][v] + sigmoidf_(gi) * tanhf(gg);
                c_reg[mt][v] = c;
                float h = sigmoidf_(go) * tanhf(c);
                if (hunit < H_) {
                    hbuf[nxt][b * KH_ + hunit] = (bf16)h;
                    out[((size_t)b * S_ + tt) * K2H_ + dir * H_ + hunit] = (bf16)h;
                }
            }
        }
        __syncthreads();
    }
}

// ---------------- emissions: out1(bf16) x lin_w + lin_b -> (B,S,T) fp32 ----------------
__global__ void emissions_kernel(const bf16* __restrict__ o1, const float* __restrict__ lw,
                                 const float* __restrict__ lb, float* __restrict__ em) {
    const int tid = blockIdx.x * blockDim.x + threadIdx.x;
    const int tt = tid & 31;
    const int r  = tid >> 5;
    if (r >= MROWS_ || tt >= T_) return;
    const bf16* row  = o1 + (size_t)r * K2H_;
    const float* wrw = lw + (size_t)tt * (2 * H_);
    float s = lb[tt];
    for (int k = 0; k < 2 * H_; ++k) s += (float)row[k] * wrw[k];
    em[(size_t)r * T_ + tt] = s;
}

// ---------------- CRF: one wave per batch row ----------------
__global__ __launch_bounds__(32)
void crf_kernel(const int* __restrict__ y, const int* __restrict__ mask,
                const float* __restrict__ em, const float* __restrict__ st,
                const float* __restrict__ en, const float* __restrict__ tr,
                float* __restrict__ part) {
    const int b = blockIdx.x, l = threadIdx.x;
    const int* yb = y + b * S_;
    const int* mb = mask + b * S_;
    const float* eb = em + (size_t)b * S_ * T_;

    // numerator partials + length
    float np = 0.0f; int lenp = 0;
    for (int s = l; s < S_; s += 32) {
        int m = mb[s];
        lenp += m;
        if (s >= 1 && m) np += tr[yb[s - 1] * T_ + yb[s]] + eb[s * T_ + yb[s]];
    }
    for (int o = 16; o; o >>= 1) { np += __shfl_down(np, o); lenp += __shfl_down(lenp, o); }
    const int len = __shfl(lenp, 0);

    // forward algorithm: lane j holds alpha[j]
    const int jj = (l < T_) ? l : 0;
    float alpha = (l < T_) ? st[l] + eb[l] : -1e30f;
    for (int s = 1; s < S_; ++s) {
        float mrun = -1e30f, srun = 0.0f;
        for (int i = 0; i < T_; ++i) {
            float v = __shfl(alpha, i) + tr[i * T_ + jj];
            if (v > mrun) { srun = srun * __expf(mrun - v) + 1.0f; mrun = v; }
            else          { srun += __expf(v - mrun); }
        }
        float nxtv = mrun + __logf(srun) + eb[s * T_ + jj];
        if (mb[s] && l < T_) alpha = nxtv;
    }
    float av = (l < T_) ? alpha + en[l] : -1e30f;
    float mx = av;
    for (int o = 16; o; o >>= 1) mx = fmaxf(mx, __shfl_down(mx, o));
    mx = __shfl(mx, 0);
    float se = __expf(av - mx);
    for (int o = 16; o; o >>= 1) se += __shfl_down(se, o);
    if (l == 0) {
        float logZ = mx + __logf(se);
        float num  = st[yb[0]] + eb[yb[0]] + np + en[yb[len - 1]];
        part[b] = num - logZ;
    }
}

__global__ void final_sum_kernel(const float* __restrict__ part, float* __restrict__ out) {
    if (threadIdx.x == 0) {
        float s = 0.0f;
        for (int i = 0; i < B_; ++i) s += part[i];   // fixed order -> deterministic
        out[0] = s;
    }
}

// ---------------- host launch ----------------

extern "C" void kernel_launch(void* const* d_in, const int* in_sizes, int n_in,
                              void* d_out, int out_size, void* d_ws, size_t ws_size,
                              hipStream_t stream) {
    (void)in_sizes; (void)n_in; (void)out_size; (void)ws_size;

    const int*   x      = (const int*)  d_in[0];
    const int*   y      = (const int*)  d_in[1];
    const int*   mask   = (const int*)  d_in[2];
    const float* emb    = (const float*)d_in[3];
    const float* wih[4] = { (const float*)d_in[4],  (const float*)d_in[7],
                            (const float*)d_in[10], (const float*)d_in[13] }; // l0f,l0b,l1f,l1b
    const float* whh[4] = { (const float*)d_in[5],  (const float*)d_in[8],
                            (const float*)d_in[11], (const float*)d_in[14] };
    const float* bia[4] = { (const float*)d_in[6],  (const float*)d_in[9],
                            (const float*)d_in[12], (const float*)d_in[15] };
    const float* lin_w  = (const float*)d_in[16];
    const float* lin_b  = (const float*)d_in[17];
    const float* start_t= (const float*)d_in[18];
    const float* end_t  = (const float*)d_in[19];
    const float* trans  = (const float*)d_in[20];
    float* out = (float*)d_out;

    // workspace carve (all blocks are multiples of 256 B)
    char* ws = (char*)d_ws;
    float* xg    = (float*)ws;  ws += (size_t)S_ * B_ * NG_ * 4;       // 218.1 MB
    bf16*  embx  = (bf16*) ws;  ws += (size_t)MROWS_ * KE_ * 2;        // 4.2 MB
    bf16*  out0  = (bf16*) ws;  ws += (size_t)MROWS_ * K2H_ * 2;       // 27.3 MB
    bf16*  out1  = (bf16*) ws;  ws += (size_t)MROWS_ * K2H_ * 2;       // 27.3 MB
    bf16*  wih0  = (bf16*) ws;  ws += (size_t)GROWS_ * KE_ * 2;
    bf16*  wih1  = (bf16*) ws;  ws += (size_t)GROWS_ * K2H_ * 2;
    bf16*  whhp[4];
    for (int i = 0; i < 4; ++i) { whhp[i] = (bf16*)ws; ws += (size_t)WHROWS_ * KH_ * 2; }
    float* bias0 = (float*)ws;  ws += NG_ * 4;
    float* bias1 = (float*)ws;  ws += NG_ * 4;
    float* emis  = (float*)ws;  ws += (size_t)MROWS_ * T_ * 4;
    float* part  = (float*)ws;  ws += 256;

    // 1) zero bf16 activation buffers (pad columns must be exactly 0 for WMMA K-padding)
    {
        long n = (long)MROWS_ * K2H_;
        int g = (int)((n + 255) / 256);
        zero_bf16_kernel<<<g, 256, 0, stream>>>(out0, n);
        zero_bf16_kernel<<<g, 256, 0, stream>>>(out1, n);
    }
    // 2) embedding gather -> bf16, K padded to 64, PAD row zeroed
    gather_emb_kernel<<<(MROWS_ * KE_) / 256, 256, 0, stream>>>(x, emb, embx);

    // 3) weight packing (fp32 -> bf16, K zero-padded; w_hh rows padded to 832)
    {
        auto g = [](long n) { return (int)((n + 255) / 256); };
        pack_bf16_kernel<<<g(800L * KE_), 256, 0, stream>>>(wih[0], wih0, 800, E_, 800, KE_, 0);
        pack_bf16_kernel<<<g(800L * KE_), 256, 0, stream>>>(wih[1], wih0, 800, E_, 800, KE_, 800);
        pack_bf16_kernel<<<g(800L * K2H_), 256, 0, stream>>>(wih[2], wih1, 800, 2 * H_, 800, K2H_, 0);
        pack_bf16_kernel<<<g(800L * K2H_), 256, 0, stream>>>(wih[3], wih1, 800, 2 * H_, 800, K2H_, 800);
        for (int i = 0; i < 4; ++i)
            pack_bf16_kernel<<<g((long)WHROWS_ * KH_), 256, 0, stream>>>(whh[i], whhp[i],
                                                                         800, H_, WHROWS_, KH_, 0);
        pack_bias_kernel<<<(NG_ + 255) / 256, 256, 0, stream>>>(bia[0], bia[1], bias0);
        pack_bias_kernel<<<(NG_ + 255) / 256, 256, 0, stream>>>(bia[2], bia[3], bias1);
    }

    dim3 ggrid(MROWS_ / 128, GROWS_ / 64);   // (256, 25)

    // 4) layer 0: gates GEMM (K=64) -> scan
    gates_gemm_kernel<<<ggrid, 256, 0, stream>>>(embx, wih0, bias0, xg, KE_, KE_ / 32);
    lstm_scan_kernel<<<2, 416, 0, stream>>>(xg, whhp[0], whhp[1], out0);

    // 5) layer 1: gates GEMM (K=416) -> scan
    gates_gemm_kernel<<<ggrid, 256, 0, stream>>>(out0, wih1, bias1, xg, K2H_, K2H_ / 32);
    lstm_scan_kernel<<<2, 416, 0, stream>>>(xg, whhp[2], whhp[3], out1);

    // 6) emissions + CRF + deterministic reduction
    emissions_kernel<<<(MROWS_ * 32) / 256, 256, 0, stream>>>(out1, lin_w, lin_b, emis);
    crf_kernel<<<B_, 32, 0, stream>>>(y, mask, emis, start_t, end_t, trans, part);
    final_sum_kernel<<<1, 1, 0, stream>>>(part, out);
}
